// STEncoder_39591008534946
// MI455X (gfx1250) — matrix operand order repaired and checked by reading, not compiled, not run on hardware
//
#include <hip/hip_runtime.h>
#include <hip/hip_bf16.h>
#include <math.h>

#define Nn 716
#define Tt 12
#define BTt 768
#define Dd 64

typedef __attribute__((ext_vector_type(16))) __bf16 v16bf;
typedef __attribute__((ext_vector_type(8)))  float  v8f;
typedef __attribute__((ext_vector_type(4)))  int    v4i;

#define AS1 __attribute__((address_space(1)))
#define AS3 __attribute__((address_space(3)))

#if defined(__has_builtin)
#  if __has_builtin(__builtin_amdgcn_global_load_async_to_lds_b128)
#    define HAS_ASYNC_LDS 1
#  endif
#endif
#ifndef HAS_ASYNC_LDS
#  define HAS_ASYNC_LDS 0
#endif

__device__ inline unsigned short f2bf(float x) {
    unsigned int u = __float_as_uint(x);
    u += 0x7FFFu + ((u >> 16) & 1u);          // round-to-nearest-even
    return (unsigned short)(u >> 16);
}

__device__ inline void wait_async() {
#if HAS_ASYNC_LDS
#  if __has_builtin(__builtin_amdgcn_s_wait_asynccnt)
    __builtin_amdgcn_s_wait_asynccnt(0);
#  else
    asm volatile("s_wait_asynccnt 0x0" ::: "memory");
#  endif
#endif
}

// ---------------------------------------------------------------------------
// Kernel 1: Aad[m][n] = softmax_row(xs xs^T)[m,n] + softmax_row(relu(E E^T))[m,n]
// xs = x[63, 6]  (flat bt index 762)
// ---------------------------------------------------------------------------
__global__ __launch_bounds__(256) void k_build_aad(const float* __restrict__ x,
                                                   const float* __restrict__ emb,
                                                   float* __restrict__ Aad) {
    const int m   = blockIdx.x;
    const int tid = threadIdx.x;
    __shared__ float xrow[64];
    __shared__ float erow[10];
    __shared__ float d1[Nn], d2[Nn];
    __shared__ float red[256];
    const float* xs = x + (size_t)762 * Nn * Dd;

    if (tid < 64) xrow[tid] = xs[(size_t)m * 64 + tid];
    if (tid < 10) erow[tid] = emb[m * 10 + tid];
    __syncthreads();

    for (int n = tid; n < Nn; n += 256) {
        const float* xr = xs + (size_t)n * 64;
        float s = 0.f;
        for (int d = 0; d < 64; ++d) s += xrow[d] * xr[d];
        d1[n] = s;
        const float* er = emb + n * 10;
        float e = 0.f;
        for (int j = 0; j < 10; ++j) e += erow[j] * er[j];
        d2[n] = e > 0.f ? e : 0.f;
    }
    __syncthreads();

    float m1 = -1e30f, m2 = -1e30f;
    for (int n = tid; n < Nn; n += 256) { m1 = fmaxf(m1, d1[n]); m2 = fmaxf(m2, d2[n]); }
    red[tid] = m1; __syncthreads();
    for (int s = 128; s > 0; s >>= 1) { if (tid < s) red[tid] = fmaxf(red[tid], red[tid + s]); __syncthreads(); }
    m1 = red[0]; __syncthreads();
    red[tid] = m2; __syncthreads();
    for (int s = 128; s > 0; s >>= 1) { if (tid < s) red[tid] = fmaxf(red[tid], red[tid + s]); __syncthreads(); }
    m2 = red[0]; __syncthreads();

    float s1 = 0.f, s2 = 0.f;
    for (int n = tid; n < Nn; n += 256) { s1 += expf(d1[n] - m1); s2 += expf(d2[n] - m2); }
    red[tid] = s1; __syncthreads();
    for (int s = 128; s > 0; s >>= 1) { if (tid < s) red[tid] += red[tid + s]; __syncthreads(); }
    s1 = red[0]; __syncthreads();
    red[tid] = s2; __syncthreads();
    for (int s = 128; s > 0; s >>= 1) { if (tid < s) red[tid] += red[tid + s]; __syncthreads(); }
    s2 = red[0]; __syncthreads();

    const float i1 = 1.f / s1, i2 = 1.f / s2;
    for (int n = tid; n < Nn; n += 256)
        Aad[(size_t)m * Nn + n] = expf(d1[n] - m1) * i1 + expf(d2[n] - m2) * i2;
}

// ---------------------------------------------------------------------------
// Kernel 2: per-row k-th largest (k = 429,358,286) via radix select on float
// bit patterns (values > 0 so uint order == float order). One wave per k.
// ---------------------------------------------------------------------------
__global__ __launch_bounds__(128) void k_kth(const float* __restrict__ Aad,
                                             float* __restrict__ kth) {
    const int m   = blockIdx.x;
    const int tid = threadIdx.x;
    __shared__ unsigned int row[Nn];
    for (int n = tid; n < Nn; n += 128) row[n] = __float_as_uint(Aad[(size_t)m * Nn + n]);
    __syncthreads();
    const int wave = tid >> 5, lane = tid & 31;
    if (wave < 3) {
        const int ks[3] = {429, 358, 286};
        const int k = ks[wave];
        unsigned int cand = 0u;
        for (int bit = 30; bit >= 0; --bit) {
            unsigned int test = cand | (1u << bit);
            int cnt = 0;
            for (int n = lane; n < Nn; n += 32) cnt += (row[n] >= test) ? 1 : 0;
            for (int off = 16; off; off >>= 1) cnt += __shfl_xor(cnt, off, 32);
            if (cnt >= k) cand = test;
        }
        if (lane == 0) kth[m * 3 + wave] = __uint_as_float(cand);
    }
}

// ---------------------------------------------------------------------------
// Kernel 3: pre-masked bf16 A matrices, zero-padded to [3][720][736]
// ---------------------------------------------------------------------------
__global__ __launch_bounds__(256) void k_abf(const float* __restrict__ Aad,
                                             const float* __restrict__ kth,
                                             unsigned short* __restrict__ Abf) {
    size_t idx = (size_t)blockIdx.x * 256 + threadIdx.x;   // < 3*720*736
    int n = (int)(idx % 736);
    size_t r = idx / 736;
    int m = (int)(r % 720);
    int i = (int)(r / 720);
    float v = 0.f;
    if (m < Nn && n < Nn) {
        float a = Aad[(size_t)m * Nn + n];
        v = (a >= kth[m * 3 + i]) ? a : 0.f;
    }
    Abf[idx] = f2bf(v);
}

// ---------------------------------------------------------------------------
// Kernel 4: fold weights.  M_i[o][d] = a_i * sum_c W_i[c][d]*Wproj[o][c] (c<32)
// outconst[o] = bproj[o] + sum_c (a1 b1 + a2 b2 + a3 b3)[c] * Wproj[o][c]
// ---------------------------------------------------------------------------
__global__ __launch_bounds__(256) void k_minit(const float* W1, const float* b1,
                                               const float* W2, const float* b2,
                                               const float* W3, const float* b3,
                                               const float* Wproj, const float* bproj,
                                               const float* a1, const float* a2, const float* a3,
                                               float* Mmat, float* outconst) {
    const int tid = threadIdx.x;
    const float av[3] = {a1[0], a2[0], a3[0]};
    const float* Ws[3] = {W1, W2, W3};
    for (int idx = tid; idx < 3 * 64 * 64; idx += 256) {
        int i = idx >> 12, o = (idx >> 6) & 63, d = idx & 63;
        const float* W = Ws[i];
        float s = 0.f;
        for (int c = 0; c < 32; ++c) s += W[c * 64 + d] * Wproj[o * 64 + c];
        Mmat[idx] = av[i] * s;
    }
    if (tid < 64) {
        float s = bproj[tid];
        for (int c = 0; c < 32; ++c) {
            float beta = av[0] * b1[c] + av[1] * b2[c] + av[2] * b3[c];
            s += beta * Wproj[tid * 64 + c];
        }
        outconst[tid] = s;
    }
}

// ---------------------------------------------------------------------------
// Kernel 5: spatial GEMM.  grid (45 m-tiles, 768 bt), block 128 (4 waves).
// Double-buffered async-LDS staging of A (pre-masked bf16) and X (f32 raw),
// bf16 WMMA 16x16x32 with 3 accumulators (one per topk mask).
// ---------------------------------------------------------------------------
__global__ __launch_bounds__(128) void k_spatial(const float* __restrict__ x,
                                                 const unsigned short* __restrict__ Abf,
                                                 const float* __restrict__ Mmat,
                                                 const float* __restrict__ outconst,
                                                 float* __restrict__ out) {
    const int mtile = blockIdx.x;       // 0..44
    const int bt    = blockIdx.y;       // 0..767
    const int m0    = mtile * 16;
    const int tid   = threadIdx.x;
    const int lane  = tid & 31;
    const int wave  = tid >> 5;         // d-tile index
    const int half  = lane >> 4;
    const int mrow  = lane & 15;
    const int d0    = wave * 16;

    __shared__ alignas(16) float          sXf[2][32][68];   // 272B rows (16B aligned)
    __shared__ alignas(16) unsigned short sA[2][3][16][40]; // 80B rows (16B aligned)
    __shared__ float                      sY[3][16][66];

    const float* xb = x + (size_t)bt * Nn * Dd;

    auto stage = [&](int c, int bu) {
        const int n0 = c * 32;
#if HAS_ASYNC_LDS
        // X chunk: 32 rows x 256B = 512 x 16B packets (4 per thread)
        #pragma unroll
        for (int q = 0; q < 4; ++q) {
            int idx = tid + 128 * q;
            int kk = idx >> 4, c4 = idx & 15;
            int nsrc = n0 + kk; nsrc = (nsrc < Nn) ? nsrc : (Nn - 1); // clamp: A pad is 0 there
            const float* g = xb + (size_t)nsrc * 64 + c4 * 4;
            __builtin_amdgcn_global_load_async_to_lds_b128(
                (AS1 v4i*)g, (AS3 v4i*)&sXf[bu][kk][c4 * 4], 0, 0);
        }
        // A chunk: 3 masks x 16 rows x 64B = 192 x 16B packets
        #pragma unroll
        for (int q = 0; q < 2; ++q) {
            int idx = tid + 128 * q;
            if (idx < 192) {
                int row = idx >> 2, c4 = idx & 3;
                int i = row >> 4, mm = row & 15;
                const unsigned short* g =
                    Abf + ((size_t)(i * 720 + m0 + mm) * 736 + n0 + c4 * 8);
                __builtin_amdgcn_global_load_async_to_lds_b128(
                    (AS1 v4i*)g, (AS3 v4i*)&sA[bu][i][mm][c4 * 8], 0, 0);
            }
        }
#else
        #pragma unroll
        for (int q = 0; q < 4; ++q) {
            int idx = tid + 128 * q;
            int kk = idx >> 4, c4 = idx & 15;
            int nsrc = n0 + kk; nsrc = (nsrc < Nn) ? nsrc : (Nn - 1);
            *reinterpret_cast<float4*>(&sXf[bu][kk][c4 * 4]) =
                *reinterpret_cast<const float4*>(xb + (size_t)nsrc * 64 + c4 * 4);
        }
        #pragma unroll
        for (int q = 0; q < 2; ++q) {
            int idx = tid + 128 * q;
            if (idx < 192) {
                int row = idx >> 2, c4 = idx & 3;
                int i = row >> 4, mm = row & 15;
                *reinterpret_cast<uint4*>(&sA[bu][i][mm][c4 * 8]) =
                    *reinterpret_cast<const uint4*>(
                        Abf + ((size_t)(i * 720 + m0 + mm) * 736 + n0 + c4 * 8));
            }
        }
#endif
    };

    v8f acc0 = {0.f,0.f,0.f,0.f,0.f,0.f,0.f,0.f};
    v8f acc1 = acc0, acc2 = acc0;

    stage(0, 0);
    wait_async();
    __syncthreads();

    for (int c = 0; c < 23; ++c) {
        const int bu = c & 1;
        if (c + 1 < 23) stage(c + 1, bu ^ 1);

        // fragments per documented 16-bit A (16x32) / B (32x16) layouts
        union { v16bf v; unsigned short h[16]; } fa0, fa1, fa2, fb;
        const int krow = mrow + (half << 4);
        #pragma unroll
        for (int r = 0; r < 8; ++r) {
            int kb = ((r >> 2) << 4) + (half << 3) + ((r & 3) << 1);
            fa0.h[2 * r]     = sA[bu][0][mrow][kb];
            fa0.h[2 * r + 1] = sA[bu][0][mrow][kb + 1];
            fa1.h[2 * r]     = sA[bu][1][mrow][kb];
            fa1.h[2 * r + 1] = sA[bu][1][mrow][kb + 1];
            fa2.h[2 * r]     = sA[bu][2][mrow][kb];
            fa2.h[2 * r + 1] = sA[bu][2][mrow][kb + 1];
            fb.h[2 * r]      = f2bf(sXf[bu][kb][d0 + mrow]);
            fb.h[2 * r + 1]  = f2bf(sXf[bu][kb + 1][d0 + mrow]);
        }
        acc0 = __builtin_amdgcn_wmma_f32_16x16x32_bf16(false, fa0.v, false, fb.v, (short)0, acc0, false, false);
        acc1 = __builtin_amdgcn_wmma_f32_16x16x32_bf16(false, fa1.v, false, fb.v, (short)0, acc1, false, false);
        acc2 = __builtin_amdgcn_wmma_f32_16x16x32_bf16(false, fa2.v, false, fb.v, (short)0, acc2, false, false);

        wait_async();       // next chunk's async copies done (per wave)
        __syncthreads();    // visible to all waves; compute of c finished
    }

    // dump accumulators (C/D layout: N=lane&15, M=r+8*half)
    const int ncol = lane & 15;
    #pragma unroll
    for (int r = 0; r < 8; ++r) {
        int mm = r + (half << 3);
        sY[0][mm][d0 + ncol] = acc0[r];
        sY[1][mm][d0 + ncol] = acc1[r];
        sY[2][mm][d0 + ncol] = acc2[r];
    }
    __syncthreads();

    // out[bt, m, o] = outconst[o] + sum_i sum_d Y_i[mm][d] * M_i[o][d]
    for (int idx = tid; idx < 16 * 64; idx += 128) {
        int mm = idx >> 6, o = idx & 63;
        int m = m0 + mm;
        if (m >= Nn) continue;
        float s = outconst[o];
        for (int i = 0; i < 3; ++i) {
            const float* Mi = Mmat + (((size_t)i << 12) + (size_t)o * 64);
            const float* Yi = &sY[i][mm][0];
            float a = 0.f;
            for (int d = 0; d < 64; ++d) a += Yi[d] * Mi[d];
            s += a;
        }
        out[((size_t)bt * Nn + m) * 64 + o] = s;
    }
}

// ---------------------------------------------------------------------------
// Kernel 6: temporal branch, one block (64 thr) per (b, n).
// DFT(7 bins) -> energy/median/mask -> low_t ; high_t = sqrt(12)*delta_t0 - low_t
// x_l/x_h = gelu(.@Wa^T + ba) + x ; q/k proj ; 2x2 tiny attentions ;
// out += ZT @ WprojH^T
// ---------------------------------------------------------------------------
__global__ __launch_bounds__(64) void k_temporal(const float* __restrict__ x,
                                                 const float* __restrict__ Wql,
                                                 const float* __restrict__ Wqh,
                                                 const float* __restrict__ Wkl,
                                                 const float* __restrict__ Wkh,
                                                 const float* __restrict__ Wa1,
                                                 const float* __restrict__ ba1,
                                                 const float* __restrict__ Wa2,
                                                 const float* __restrict__ ba2,
                                                 const float* __restrict__ Wproj,
                                                 const float* __restrict__ thr_p,
                                                 float* __restrict__ out) {
    const int n   = blockIdx.x;
    const int b   = blockIdx.y;
    const int tid = threadIdx.x;
    const float RS12 = 0.28867513459481288f;   // 1/sqrt(12)
    const float SQ12 = 3.4641016151377544f;    // sqrt(12)
    const float SCALE = 0.35355339059327373f;  // 8^-0.5

    __shared__ float xt[12][64];
    __shared__ float Fr[7][64], Fi[7][64];
    __shared__ float low[12][64], high[12][64];
    __shared__ float xl[12][64], xh[12][64];
    __shared__ float qh[12][16], ql[12][16], kh[12][16], kl[12][16];
    __shared__ float ZT[12][32];
    __shared__ float energy[7], maskf[7];
    __shared__ float ct[7][12], st[7][12];
    __shared__ float red[64];

    for (int idx = tid; idx < 768; idx += 64) {
        int t = idx >> 6, d = idx & 63;
        xt[t][d] = x[(((size_t)b * 12 + t) * Nn + n) * 64 + d];
    }
    for (int idx = tid; idx < 84; idx += 64) {
        int f = idx / 12, t = idx % 12;
        float ang = 0.52359877559829887f * (float)(f * t);  // 2*pi*f*t/12
        ct[f][t] = cosf(ang);
        st[f][t] = sinf(ang);
    }
    __syncthreads();

    for (int idx = tid; idx < 448; idx += 64) {
        int f = idx >> 6, d = idx & 63;
        float sr = 0.f, si = 0.f;
        for (int t = 0; t < 12; ++t) { sr += ct[f][t] * xt[t][d]; si += st[f][t] * xt[t][d]; }
        Fr[f][d] = sr * RS12;
        Fi[f][d] = -si * RS12;
    }
    __syncthreads();

    for (int f = 0; f < 7; ++f) {
        float v = Fr[f][tid] * Fr[f][tid] + Fi[f][tid] * Fi[f][tid];
        red[tid] = v; __syncthreads();
        for (int s = 32; s > 0; s >>= 1) { if (tid < s) red[tid] += red[tid + s]; __syncthreads(); }
        if (tid == 0) energy[f] = red[0];
        __syncthreads();
    }
    if (tid == 0) {
        float e[7];
        for (int f = 0; f < 7; ++f) e[f] = energy[f];
        for (int i = 1; i < 7; ++i) {
            float v = e[i]; int j = i - 1;
            while (j >= 0 && e[j] > v) { e[j + 1] = e[j]; --j; }
            e[j + 1] = v;
        }
        float inv = 1.f / (e[3] + 1e-6f);       // median of 7
        float thrv = thr_p[0];
        for (int f = 0; f < 7; ++f) maskf[f] = (energy[f] * inv > thrv) ? 1.f : 0.f;
    }
    __syncthreads();

    for (int idx = tid; idx < 768; idx += 64) {
        int t = idx >> 6, d = idx & 63;
        float s = maskf[0] * Fr[0][d];
        for (int f = 1; f < 6; ++f)
            s += 2.f * maskf[f] * (Fr[f][d] * ct[f][t] - Fi[f][d] * st[f][t]);
        s += maskf[6] * Fr[6][d] * ct[6][t];
        float lv = s * RS12;
        low[t][d]  = lv;
        high[t][d] = (t == 0 ? SQ12 : 0.f) - lv;
    }
    __syncthreads();

    for (int idx = tid; idx < 768; idx += 64) {
        int t = idx >> 6, d = idx & 63;
        float z1 = ba1[d], z2 = ba2[d];
        const float* w1 = Wa1 + d * 64;
        const float* w2 = Wa2 + d * 64;
        for (int e = 0; e < 64; ++e) { z1 += low[t][e] * w1[e]; z2 += high[t][e] * w2[e]; }
        float g1 = 0.5f * z1 * (1.f + erff(z1 * 0.70710678118654752f));
        float g2 = 0.5f * z2 * (1.f + erff(z2 * 0.70710678118654752f));
        xl[t][d] = g1 + xt[t][d];
        xh[t][d] = g2 + xt[t][d];
    }
    __syncthreads();

    for (int idx = tid; idx < 768; idx += 64) {
        int unit = idx / 192, rem = idx % 192;
        int t = rem >> 4, c = rem & 15;
        const float* W = (unit == 0) ? Wql : (unit == 1) ? Wqh : (unit == 2) ? Wkl : Wkh;
        const float (*src)[64] = (unit <= 1) ? xt : (unit == 2) ? xl : xh;
        float s = 0.f;
        const float* w = W + c * 64;
        for (int d = 0; d < 64; ++d) s += src[t][d] * w[d];
        if (unit == 0) ql[t][c] = s;
        else if (unit == 1) qh[t][c] = s;
        else if (unit == 2) kl[t][c] = s;
        else kh[t][c] = s;
    }
    __syncthreads();

    if (tid < 48) {
        int p = tid / 24, rem = tid % 24;
        int h = rem / 12, q = rem % 12;
        const float (*Q)[16] = p ? ql : qh;
        const float (*K)[16] = p ? kl : kh;
        float sc[12], mx = -1e30f;
        for (int k = 0; k < 12; ++k) {
            float s = 0.f;
            for (int dd = 0; dd < 8; ++dd) s += Q[q][h * 8 + dd] * K[k][h * 8 + dd];
            sc[k] = s * SCALE;
            mx = fmaxf(mx, sc[k]);
        }
        float sum = 0.f;
        for (int k = 0; k < 12; ++k) { sc[k] = expf(sc[k] - mx); sum += sc[k]; }
        float inv = 1.f / sum;
        for (int dd = 0; dd < 8; ++dd) {
            float o = 0.f;
            for (int k = 0; k < 12; ++k) o += sc[k] * K[k][h * 8 + dd];
            ZT[q][p * 16 + h * 8 + dd] = o * inv;
        }
    }
    __syncthreads();

    for (int idx = tid; idx < 768; idx += 64) {
        int t = idx >> 6, o = idx & 63;
        float v = 0.f;
        const float* w = Wproj + o * 64 + 32;
        for (int c = 0; c < 32; ++c) v += ZT[t][c] * w[c];
        size_t oi = (((size_t)b * 12 + t) * Nn + n) * 64 + o;
        out[oi] += v;
    }
}

// ---------------------------------------------------------------------------
extern "C" void kernel_launch(void* const* d_in, const int* in_sizes, int n_in,
                              void* d_out, int out_size, void* d_ws, size_t ws_size,
                              hipStream_t stream) {
    (void)in_sizes; (void)n_in; (void)out_size; (void)ws_size;
    const float* x     = (const float*)d_in[0];
    const float* emb   = (const float*)d_in[3];
    const float* W1    = (const float*)d_in[4];
    const float* b1    = (const float*)d_in[5];
    const float* W2    = (const float*)d_in[6];
    const float* b2    = (const float*)d_in[7];
    const float* W3    = (const float*)d_in[8];
    const float* b3    = (const float*)d_in[9];
    const float* Wproj = (const float*)d_in[10];
    const float* bproj = (const float*)d_in[11];
    const float* Wql   = (const float*)d_in[12];
    const float* Wqh   = (const float*)d_in[13];
    const float* Wkl   = (const float*)d_in[14];
    const float* Wkh   = (const float*)d_in[15];
    const float* Wa1   = (const float*)d_in[16];
    const float* ba1   = (const float*)d_in[17];
    const float* Wa2   = (const float*)d_in[18];
    const float* ba2   = (const float*)d_in[19];
    const float* a1    = (const float*)d_in[20];
    const float* a2    = (const float*)d_in[21];
    const float* a3    = (const float*)d_in[22];
    const float* thr   = (const float*)d_in[23];

    float* ws   = (float*)d_ws;
    float* Aad  = ws;                                  // 716*716 f32
    float* kthv = Aad + (size_t)Nn * Nn;               // 716*3 f32
    float* Mmat = kthv + Nn * 3;                       // 3*64*64 f32
    float* outc = Mmat + 3 * 64 * 64;                  // 64 f32
    unsigned short* Abf = (unsigned short*)(outc + 64);// [3][720][736] bf16 (16B aligned)
    float* out  = (float*)d_out;

    k_build_aad<<<Nn, 256, 0, stream>>>(x, emb, Aad);
    k_kth<<<Nn, 128, 0, stream>>>(Aad, kthv);
    k_abf<<<(3 * 720 * 736) / 256, 256, 0, stream>>>(Aad, kthv, Abf);
    k_minit<<<1, 256, 0, stream>>>(W1, b1, W2, b2, W3, b3, Wproj, bproj, a1, a2, a3, Mmat, outc);
    k_spatial<<<dim3(45, BTt), 128, 0, stream>>>(x, Abf, Mmat, outc, out);
    k_temporal<<<dim3(Nn, 64), 64, 0, stream>>>(x, Wql, Wqh, Wkl, Wkh,
                                                Wa1, ba1, Wa2, ba2, Wproj, thr, out);
}
